// GATLayer_39539468927411
// MI455X (gfx1250) — matrix operand order repaired and checked by reading, not compiled, not run on hardware
//
#include <hip/hip_runtime.h>
#include <hip/hip_bf16.h>
#include <math.h>

typedef __attribute__((ext_vector_type(2)))  float  v2f;
typedef __attribute__((ext_vector_type(8)))  float  v8f;
typedef __attribute__((ext_vector_type(16))) __bf16 v16bf;
typedef __attribute__((ext_vector_type(4)))  int    i32x4;
typedef __attribute__((ext_vector_type(8)))  int    i32x8;
typedef __attribute__((ext_vector_type(4)))  unsigned int u32x4;

#define N_NODES 4096
#define F_IN    512
#define F_OUT   64
#define H_HEADS 8
#define ALPHA   0.2f
#define MROW    34   // LDS dwords per mask row: 32 data + 2 TDM pad (conflict-free pitch)

// ---------------------------------------------------------------------------
// Kernel A: Wh[h] = h(4096x512) x W[h](512x64) + bW  via V_WMMA_F32_16X16X4_F32
// Writes Wh f32 [H][N][64] and WhT bf16 [H][64][N] (transposed for B-operand).
// ---------------------------------------------------------------------------
__global__ __launch_bounds__(256) void gat_wh_kernel(
    const float* __restrict__ hmat, const float* __restrict__ W,
    const float* __restrict__ bW,
    float* __restrict__ Wh, __bf16* __restrict__ WhT)
{
    const int lane = threadIdx.x & 31;
    const int wave = threadIdx.x >> 5;
    const int head = blockIdx.x >> 5;                 // 8 heads x 32 blocks
    const int n0   = ((blockIdx.x & 31) * 8 + wave) * 16;
    const int row  = lane & 15;                        // M (A) / N (B,C)
    const int hi   = lane >> 4;

    const float* hrow = hmat + (size_t)(n0 + row) * F_IN;
    const float* Wmat = W + (size_t)head * F_IN * F_OUT;

    v8f acc[4] = {};

    for (int k0 = 0; k0 < F_IN; k0 += 4) {
        const int kb = k0 + 2 * hi;                    // f32 A/B K-split per lane half
        const v2f a = *(const v2f*)(hrow + kb);        // A: h[M][kb], h[M][kb+1]
        const float* w0 = Wmat + (size_t)kb * F_OUT + row;
#pragma unroll
        for (int t = 0; t < 4; ++t) {
            v2f b;
            b.x = w0[t * 16];                          // W[kb  ][t*16+row]
            b.y = w0[F_OUT + t * 16];                  // W[kb+1][t*16+row]
            acc[t] = __builtin_amdgcn_wmma_f32_16x16x4_f32(
                false, a, false, b, (short)0, acc[t], false, false);
        }
    }

    const float* bWh = bW + head * F_OUT;
#pragma unroll
    for (int t = 0; t < 4; ++t) {
        const float bias = bWh[t * 16 + row];
#pragma unroll
        for (int r = 0; r < 8; ++r) {
            const int n = n0 + r + 8 * hi;             // C layout: M = r + 8*hi
            const float v = acc[t][r] + bias;
            Wh[((size_t)head * N_NODES + n) * F_OUT + t * 16 + row] = v;
            WhT[((size_t)head * F_OUT + t * 16 + row) * N_NODES + n] = (__bf16)v;
        }
    }
}

// ---------------------------------------------------------------------------
// Kernel B: el[h][n] = Wh[h][n]·a_l[h],  er likewise. One thread per (h,n).
// ---------------------------------------------------------------------------
__global__ __launch_bounds__(256) void gat_elr_kernel(
    const float* __restrict__ Wh, const float* __restrict__ a_l,
    const float* __restrict__ a_r, float* __restrict__ el, float* __restrict__ er)
{
    const int head = blockIdx.y;
    const int n = blockIdx.x * 256 + threadIdx.x;
    const float* wr = Wh + ((size_t)head * N_NODES + n) * F_OUT;
    const float* al = a_l + head * F_OUT;
    const float* ar = a_r + head * F_OUT;
    float sl = 0.f, sr = 0.f;
#pragma unroll 8
    for (int o = 0; o < F_OUT; ++o) {
        const float v = wr[o];
        sl += v * al[o];
        sr += v * ar[o];
    }
    el[head * N_NODES + n] = sl;
    er[head * N_NODES + n] = sr;
}

// ---------------------------------------------------------------------------
// Kernel C: per-head logit upper bound m_h = LRelu(max(el)+max(er)+bA).
// LeakyReLU is monotone, so this bounds every e[i,j]; subtracting a per-head
// constant leaves softmax exact while making all exp args <= 0.
// ---------------------------------------------------------------------------
__global__ __launch_bounds__(256) void gat_rowmax_kernel(
    const float* __restrict__ el, const float* __restrict__ er,
    const float* __restrict__ bA, float* __restrict__ mhead)
{
    __shared__ float s1[256], s2[256];
    const int head = blockIdx.x;
    const int tid = threadIdx.x;
    float m1 = -3.4e38f, m2 = -3.4e38f;
    for (int i = tid; i < N_NODES; i += 256) {
        m1 = fmaxf(m1, el[head * N_NODES + i]);
        m2 = fmaxf(m2, er[head * N_NODES + i]);
    }
    s1[tid] = m1; s2[tid] = m2;
    __syncthreads();
    for (int s = 128; s > 0; s >>= 1) {
        if (tid < s) {
            s1[tid] = fmaxf(s1[tid], s1[tid + s]);
            s2[tid] = fmaxf(s2[tid], s2[tid + s]);
        }
        __syncthreads();
    }
    if (tid == 0) {
        const float x = s1[0] + s2[0] + bA[head];
        mhead[head] = x > 0.f ? x : ALPHA * x;
    }
}

// ---------------------------------------------------------------------------
// TDM: DMA one 16x32 int32 mask tile (row stride 4096) into LDS with 2-dword
// padding after every 32 dwords -> 34-dword LDS row pitch.
// D# packing per CDNA5 ISA ch.8 (group0: count/lds_addr/global_addr/type=2;
// group1: data_size=4B, pad_interval=32dw, pad_amount=2dw, tensor/tile dims).
// clang-23 toolchain: 6-arg builtin (g0 u32x4, g1 i32x8, g2/g3 i32x4, g4 i32x8, cpol).
// ---------------------------------------------------------------------------
__device__ __forceinline__ void tdm_load_mask_tile(const int* gsrc, unsigned lds_off)
{
    const unsigned long long ga = (unsigned long long)(size_t)gsrc;
    u32x4 g0;
    g0.x = 1u;                                            // count=1, user mode
    g0.y = lds_off;                                       // lds_addr (bytes)
    g0.z = (unsigned)ga;                                  // global_addr[31:0]
    g0.w = (unsigned)((ga >> 32) & 0x1FFFFFFu) | (2u << 30); // addr[56:32] | type=2
    i32x8 g1;
    g1[0] = (2 << 16) | (1 << 20) | (4 << 22) | (1 << 25); // data_size=4B, pad_en,
                                                           // interval=32dw, amount=2dw
    g1[1] = (int)((N_NODES & 0xFFFF) << 16);               // tensor_dim0 lo16 (4096)
    g1[2] = (int)((N_NODES >> 16) | ((N_NODES & 0xFFFF) << 16)); // dim0 hi | dim1 lo
    g1[3] = (int)((N_NODES >> 16) | (32 << 16));           // dim1 hi | tile_dim0=32
    g1[4] = 16;                                            // tile_dim1=16, tile_dim2=0
    g1[5] = N_NODES;                                       // tensor_dim0_stride lo32
    g1[6] = 0;                                             // stride hi | dim1_stride lo
    g1[7] = 0;
    const i32x4 z4 = {0, 0, 0, 0};
    const i32x8 z8 = {0, 0, 0, 0, 0, 0, 0, 0};
    __builtin_amdgcn_tensor_load_to_lds(g0, g1, z4, z4, z8, 0);
}

// ---------------------------------------------------------------------------
// Kernel D: fused masked-softmax + P·Wh via V_WMMA_F32_16X16X32_BF16.
// Mask tiles arrive by double-buffered TENSOR_LOAD_TO_LDS (s_wait_tensorcnt),
// er is LDS-resident, P is built directly in the A-operand register layout:
// element e of lane L -> M = L&15, K = e + 16*(L>=16) (same K order as B).
// B-operand from WhT: lane L, elem e -> o = t*16+(L&15), K = e+16*(L>=16).
// ---------------------------------------------------------------------------
__global__ __launch_bounds__(256) void gat_attn_kernel(
    const int* __restrict__ mask, const __bf16* __restrict__ WhT,
    const float* __restrict__ el, const float* __restrict__ er,
    const float* __restrict__ bA, const float* __restrict__ mhead,
    float* __restrict__ out)
{
    __shared__ int   msk[8][2][16 * MROW];   // 34816 B, TDM double buffers
    __shared__ float ers[N_NODES];           // 16 KB, per-head er
    __shared__ float zsh[8][16];

    const int lane = threadIdx.x & 31;
    const int wave = threadIdx.x >> 5;
    const int head = blockIdx.y;
    const int i0   = blockIdx.x * 128 + wave * 16;
    const int row  = lane & 15;
    const int hi   = lane >> 4;

    const float* erh = er + head * N_NODES;
    for (int i = threadIdx.x; i < N_NODES; i += 256) ers[i] = erh[i];
    __syncthreads();

    const float elr = el[head * N_NODES + i0 + row];
    const float bAh = bA[head];
    const float mh  = mhead[head];
    const int*    mbase = mask + (size_t)i0 * N_NODES;   // wave's 16 mask rows
    const __bf16* bt    = WhT + (size_t)head * F_OUT * N_NODES;

    const unsigned loff0 = (unsigned)(size_t)&msk[wave][0][0];
    const unsigned loff1 = (unsigned)(size_t)&msk[wave][1][0];

    v8f acc[4] = {};
    float z = 0.f;

    tdm_load_mask_tile(mbase, loff0);                    // prime buffer 0

    for (int it = 0; it < N_NODES / 32; ++it) {
        const int j0 = it * 32;
        if (it + 1 < N_NODES / 32) {                      // prefetch next tile
            tdm_load_mask_tile(mbase + (j0 + 32), (it & 1) ? loff0 : loff1);
            __builtin_amdgcn_s_wait_tensorcnt(1);         // current tile landed
        } else {
            __builtin_amdgcn_s_wait_tensorcnt(0);
        }
        __asm__ volatile("" ::: "memory");                // TDM wrote LDS: no hoisting

        const int* mb = &msk[wave][it & 1][0] + row * MROW;

        v16bf aP;
#pragma unroll
        for (int m = 0; m < 8; ++m) {
            const int Kp = 2 * m + hi * 16;
            const int2   mk  = *(const int2*)(mb + Kp);
            const float2 erp = *(const float2*)(&ers[j0 + Kp]);
            float x0 = elr + erp.x + bAh; x0 = x0 > 0.f ? x0 : ALPHA * x0;
            float x1 = elr + erp.y + bAh; x1 = x1 > 0.f ? x1 : ALPHA * x1;
            const float p0 = mk.x ? __expf(x0 - mh) : 0.f;
            const float p1 = mk.y ? __expf(x1 - mh) : 0.f;
            z += p0 + p1;
            aP[2 * m]     = (__bf16)p0;
            aP[2 * m + 1] = (__bf16)p1;
        }
#pragma unroll
        for (int t = 0; t < 4; ++t) {
            const v16bf b = *(const v16bf*)(bt + ((size_t)(t * 16 + row)) * N_NODES
                                               + j0 + hi * 16);
            acc[t] = __builtin_amdgcn_wmma_f32_16x16x32_bf16(
                false, aP, false, b, (short)0, acc[t], false, false);
        }
    }

    // Combine the two K-halves of each row's sum, publish per-wave row Z.
    z += __shfl_xor(z, 16, 32);
    if (lane < 16) zsh[wave][lane] = z;   // lane L holds row L for L<16
    __syncthreads();

    float rinv[8];
#pragma unroll
    for (int r = 0; r < 8; ++r) {
        const float Z = zsh[wave][r + 8 * hi];   // C layout: M = r + 8*hi
        rinv[r] = 1.f / fmaxf(Z, 1e-30f);
    }

#pragma unroll
    for (int t = 0; t < 4; ++t) {
#pragma unroll
        for (int r = 0; r < 8; ++r) {
            float v = acc[t][r] * rinv[r];
            v = v > 0.f ? v : expm1f(v);         // ELU (alpha=1)
            out[(size_t)(i0 + r + 8 * hi) * (H_HEADS * F_OUT)
                + head * F_OUT + t * 16 + row] = v;
        }
    }
}

// ---------------------------------------------------------------------------
extern "C" void kernel_launch(void* const* d_in, const int* in_sizes, int n_in,
                              void* d_out, int out_size, void* d_ws, size_t ws_size,
                              hipStream_t stream) {
    (void)in_sizes; (void)n_in; (void)out_size; (void)ws_size;
    const float* h    = (const float*)d_in[0];
    const int*   mask = (const int*)d_in[1];
    const float* W    = (const float*)d_in[2];
    const float* bW   = (const float*)d_in[3];
    const float* a_l  = (const float*)d_in[4];
    const float* a_r  = (const float*)d_in[5];
    const float* bA   = (const float*)d_in[6];
    float* out = (float*)d_out;

    char* ws = (char*)d_ws;
    float*  Wh  = (float*)ws;                                   // 8 MB
    __bf16* WhT = (__bf16*)(ws + (size_t)8 * 1024 * 1024);      // 4 MB
    float*  el  = (float*)(ws + (size_t)12 * 1024 * 1024);      // 128 KB
    float*  er  = el + H_HEADS * N_NODES;                       // 128 KB
    float*  mh  = er + H_HEADS * N_NODES;                       // 32 B

    gat_wh_kernel<<<256, 256, 0, stream>>>(h, W, bW, Wh, WhT);
    gat_elr_kernel<<<dim3(16, 8), 256, 0, stream>>>(Wh, a_l, a_r, el, er);
    gat_rowmax_kernel<<<8, 256, 0, stream>>>(el, er, bA, mh);
    gat_attn_kernel<<<dim3(32, 8), 256, 0, stream>>>(mask, WhT, el, er, bA, mh, out);
}
